// SimpleMamba2_87256555586264
// MI455X (gfx1250) — matrix operand order repaired and checked by reading, not compiled, not run on hardware
//
#include <hip/hip_runtime.h>
#include <hip/hip_bf16.h>
#include <math.h>

#define D_MODEL 1024
#define D_STATE 16
#define D_INNER 2048
#define BATCH 2
#define SEQ 1024
#define NTOK (BATCH*SEQ)
#define LN_EPS 1e-5f

typedef __bf16 bf16;
typedef __attribute__((ext_vector_type(16))) __bf16 v16bf;
typedef __attribute__((ext_vector_type(8)))  float  v8f;

// ---------------- LayerNorm: one block per token ----------------
__global__ __launch_bounds__(256) void k_ln(const float* __restrict__ x,
                                            const float* __restrict__ gamma,
                                            const float* __restrict__ beta,
                                            float* __restrict__ xn) {
    int tok = blockIdx.x;
    int tid = threadIdx.x;
    const float* xr = x + (size_t)tok * D_MODEL;
    float v[4];
    float s = 0.f, s2 = 0.f;
#pragma unroll
    for (int i = 0; i < 4; ++i) {
        float f = xr[tid + 256 * i];
        v[i] = f; s += f; s2 += f * f;
    }
    __shared__ float r1[256], r2[256];
    r1[tid] = s; r2[tid] = s2;
    __syncthreads();
    for (int off = 128; off > 0; off >>= 1) {
        if (tid < off) { r1[tid] += r1[tid + off]; r2[tid] += r2[tid + off]; }
        __syncthreads();
    }
    float mu  = r1[0] * (1.f / D_MODEL);
    float var = r2[0] * (1.f / D_MODEL) - mu * mu;
    float inv = rsqrtf(var + LN_EPS);
#pragma unroll
    for (int i = 0; i < 4; ++i) {
        int c = tid + 256 * i;
        xn[(size_t)tok * D_MODEL + c] = (v[i] - mu) * inv * gamma[c] + beta[c];
    }
}

// ------- Pack fp32 matrix rows into wave32 WMMA A-fragments (hi/lo bf16 split) -------
// A fragment layout (16-bit A 16x32): lane m=lane&15; VGPR v holds K pair at
// kb = 32*kt + 16*(v/4) + 8*(lane/16) + 2*(v&3)
__global__ __launch_bounds__(256) void k_pack_a(const float* __restrict__ X,
                                                bf16* __restrict__ Ahi, bf16* __restrict__ Alo,
                                                int M, int K) {
    int KT = K >> 5;
    int total = (M >> 4) * KT * 32;
    int fl = blockIdx.x * 256 + threadIdx.x;
    if (fl >= total) return;
    int lane = fl & 31;
    int kt   = (fl >> 5) % KT;
    int mt   = fl / (32 * KT);
    int m    = mt * 16 + (lane & 15);
    const float* row = X + (size_t)m * K;
    v16bf hv, lv;
#pragma unroll
    for (int v = 0; v < 8; ++v) {
        int kb = kt * 32 + ((v >> 2) << 4) + ((lane >> 4) << 3) + ((v & 3) << 1);
        float f0 = row[kb], f1 = row[kb + 1];
        bf16 h0 = (bf16)f0, h1 = (bf16)f1;
        hv[2 * v]     = h0;
        hv[2 * v + 1] = h1;
        lv[2 * v]     = (bf16)(f0 - (float)h0);
        lv[2 * v + 1] = (bf16)(f1 - (float)h1);
    }
    *(v16bf*)(Ahi + (size_t)fl * 16) = hv;
    *(v16bf*)(Alo + (size_t)fl * 16) = lv;
}

// ------- Pack fp32 KxN weight into wave32 WMMA B-fragments (hi/lo bf16 split) -------
// B fragment layout (16-bit B 32x16): n=lane&15; VGPR v holds K pair at
// kb = 32*kt + 16*(v/4) + 8*(lane/16) + 2*(v&3)
__global__ __launch_bounds__(256) void k_pack_w(const float* __restrict__ W,
                                                bf16* __restrict__ Bhi, bf16* __restrict__ Blo,
                                                int K, int N, int ldw, int col0) {
    int NT = N >> 4;
    int total = (K >> 5) * NT * 32;
    int fl = blockIdx.x * 256 + threadIdx.x;
    if (fl >= total) return;
    int lane = fl & 31;
    int nt   = (fl >> 5) % NT;
    int kt   = fl / (32 * NT);
    int n    = nt * 16 + (lane & 15);
    v16bf hv, lv;
#pragma unroll
    for (int v = 0; v < 8; ++v) {
        int kb = kt * 32 + ((v >> 2) << 4) + ((lane >> 4) << 3) + ((v & 3) << 1);
        float f0 = W[(size_t)kb * ldw + col0 + n];
        float f1 = W[(size_t)(kb + 1) * ldw + col0 + n];
        bf16 h0 = (bf16)f0, h1 = (bf16)f1;
        hv[2 * v]     = h0;
        hv[2 * v + 1] = h1;
        lv[2 * v]     = (bf16)(f0 - (float)h0);
        lv[2 * v + 1] = (bf16)(f1 - (float)h1);
    }
    *(v16bf*)(Bhi + (size_t)fl * 16) = hv;
    *(v16bf*)(Blo + (size_t)fl * 16) = lv;
}

// ---- column-sum of W_in[:, 2048:4096] -> wsum[1024]; bias sum -> wsum[1024] slot ----
__global__ __launch_bounds__(256) void k_reduce_win(const float* __restrict__ W_in,
                                                    const float* __restrict__ b_in,
                                                    float* __restrict__ wsum) {
    int k = blockIdx.x * 256 + threadIdx.x;
    if (k < D_MODEL) {
        const float* r = W_in + (size_t)k * (2 * D_INNER) + D_INNER;
        float s = 0.f;
        for (int j = 0; j < D_INNER; ++j) s += r[j];
        wsum[k] = s;
    }
    if (k == 0) {
        float s = 0.f;
        for (int j = 0; j < D_INNER; ++j) s += b_in[D_INNER + j];
        wsum[D_MODEL] = s;
    }
}

// ---- small projections per token: B(16), C(16), x_sum(1) ----
__global__ __launch_bounds__(64) void k_bcx(const float* __restrict__ xn,
                                            const float* __restrict__ W_B, const float* __restrict__ b_B,
                                            const float* __restrict__ W_C, const float* __restrict__ b_C,
                                            const float* __restrict__ wsum,
                                            float* __restrict__ Bm, float* __restrict__ Cm,
                                            float* __restrict__ xsum) {
    int t = blockIdx.x;
    int j = threadIdx.x;
    if (j >= 33) return;
    const float* xr = xn + (size_t)t * D_MODEL;
    float acc = 0.f;
    if (j < 16) {
        for (int k = 0; k < D_MODEL; ++k) acc += xr[k] * W_B[k * 16 + j];
        Bm[t * 16 + j] = acc + b_B[j];
    } else if (j < 32) {
        int jj = j - 16;
        for (int k = 0; k < D_MODEL; ++k) acc += xr[k] * W_C[k * 16 + jj];
        Cm[t * 16 + jj] = acc + b_C[jj];
    } else {
        for (int k = 0; k < D_MODEL; ++k) acc += xr[k] * wsum[k];
        xsum[t] = acc + wsum[D_MODEL];
    }
}

// ---- bf16x3 split-precision WMMA GEMM: D[M,N] = A[M,K] * B[K,N] (+bias)(+resid) ----
// one wave computes a 32x64 tile (8 accumulators); 24 v_wmma per 32-wide K step.
// Fragment pointers are loop-carried with constant strides; the 4 B n-tiles of a
// group sit at fixed 1KB offsets so they fold into load instruction offsets.
__global__ __launch_bounds__(256) void k_gemm(const bf16* __restrict__ Ahi, const bf16* __restrict__ Alo,
                                              const bf16* __restrict__ Bhi, const bf16* __restrict__ Blo,
                                              float* __restrict__ D, int M, int N, int K,
                                              const float* __restrict__ bias,
                                              const float* __restrict__ resid) {
    int lane = threadIdx.x & 31;
    int wave = blockIdx.x * (blockDim.x >> 5) + (threadIdx.x >> 5);
    int KT = K >> 5, NT = N >> 4;
    int MT2 = M >> 5;                     // 32-row tile pairs
    int NG  = NT >> 2;                    // groups of 4 n-tiles (64 cols)
    int mtp = wave / NG, ng = wave % NG;
    if (mtp >= MT2) return;
    int mt0 = mtp * 2;

    v8f acc[8];
#pragma unroll
    for (int j = 0; j < 8; ++j) acc[j] = (v8f){0.f,0.f,0.f,0.f,0.f,0.f,0.f,0.f};

    // fragment size = 32 lanes * 16 bf16 = 512 elements (1KB)
    const size_t FR = 512;
    const size_t aStep = FR;              // A: kt-major within an m-tile row
    const size_t bStep = (size_t)NT * FR; // B: kt-major over all n-tiles
    const size_t aOff1 = (size_t)KT * FR; // second m-tile

    const bf16* pAh = Ahi + (size_t)mt0 * KT * FR + (size_t)lane * 16;
    const bf16* pAl = Alo + (size_t)mt0 * KT * FR + (size_t)lane * 16;
    const bf16* pBh = Bhi + (size_t)(ng * 4) * FR + (size_t)lane * 16;
    const bf16* pBl = Blo + (size_t)(ng * 4) * FR + (size_t)lane * 16;

    for (int kt = 0; kt < KT; ++kt) {
        v16bf a0h = *(const v16bf*)(pAh);
        v16bf a0l = *(const v16bf*)(pAl);
        v16bf a1h = *(const v16bf*)(pAh + aOff1);
        v16bf a1l = *(const v16bf*)(pAl + aOff1);
#pragma unroll
        for (int j = 0; j < 4; ++j) {
            v16bf bh = *(const v16bf*)(pBh + (size_t)j * FR);
            v16bf bl = *(const v16bf*)(pBl + (size_t)j * FR);
            acc[j]     = __builtin_amdgcn_wmma_f32_16x16x32_bf16(false, a0h, false, bh, (short)0, acc[j],     false, false);
            acc[j]     = __builtin_amdgcn_wmma_f32_16x16x32_bf16(false, a0h, false, bl, (short)0, acc[j],     false, false);
            acc[j]     = __builtin_amdgcn_wmma_f32_16x16x32_bf16(false, a0l, false, bh, (short)0, acc[j],     false, false);
            acc[4 + j] = __builtin_amdgcn_wmma_f32_16x16x32_bf16(false, a1h, false, bh, (short)0, acc[4 + j], false, false);
            acc[4 + j] = __builtin_amdgcn_wmma_f32_16x16x32_bf16(false, a1h, false, bl, (short)0, acc[4 + j], false, false);
            acc[4 + j] = __builtin_amdgcn_wmma_f32_16x16x32_bf16(false, a1l, false, bh, (short)0, acc[4 + j], false, false);
        }
        pAh += aStep; pAl += aStep;
        pBh += bStep; pBl += bStep;
    }

    // C/D layout: row = v + 8*(lane/16), col = lane&15
    int cl = lane & 15;
    int rhalf = (lane >> 4) << 3;
#pragma unroll
    for (int g = 0; g < 2; ++g) {
        int rbase = (mt0 + g) * 16 + rhalf;
#pragma unroll
        for (int j = 0; j < 4; ++j) {
            int col = (ng * 4 + j) * 16 + cl;
            float bv = bias ? bias[col] : 0.f;
#pragma unroll
            for (int v = 0; v < 8; ++v) {
                int row = rbase + v;
                float val = acc[g * 4 + j][v] + bv;
                if (resid) val += resid[(size_t)row * N + col];
                D[(size_t)row * N + col] = val;
            }
        }
    }
}

// ---- selective scan: 1 thread per (batch, d) channel; h[16] in VGPRs ----
// fused softplus(dt) on input and SiLU(z) gate on output; B/C/xsum staged in LDS
__global__ __launch_bounds__(256) void k_scan(const float* __restrict__ dtpre,
                                              const float* __restrict__ z,
                                              const float* __restrict__ Bm,
                                              const float* __restrict__ Cm,
                                              const float* __restrict__ xsum,
                                              const float* __restrict__ A,
                                              float* __restrict__ Y) {
    int b   = blockIdx.y;
    int tid = threadIdx.x;
    int d   = blockIdx.x * 256 + tid;
    __shared__ float sB[128][16];
    __shared__ float sC[128][16];
    __shared__ float sx[128];
    float h[16], a[16];
#pragma unroll
    for (int s = 0; s < 16; ++s) { h[s] = 0.f; a[s] = A[d * 16 + s]; }

    for (int chunk = 0; chunk < 8; ++chunk) {
        int t0 = chunk * 128;
        __syncthreads();
        size_t base = ((size_t)(b * SEQ + t0)) << 4;
        for (int e = tid; e < 128 * 16; e += 256) {
            sB[e >> 4][e & 15] = Bm[base + e];
            sC[e >> 4][e & 15] = Cm[base + e];
        }
        for (int e = tid; e < 128; e += 256) sx[e] = xsum[b * SEQ + t0 + e];
        __syncthreads();
        for (int tl = 0; tl < 128; ++tl) {
            int t = t0 + tl;
            size_t idx = ((size_t)(b * SEQ + t)) * D_INNER + d;
            float dtp = dtpre[idx];
            float zv  = z[idx];
            float sp  = fmaxf(dtp, 0.f) + log1pf(expf(-fabsf(dtp)));   // softplus
            float cf  = sp * sx[tl];
            float y   = 0.f;
#pragma unroll
            for (int s = 0; s < 16; ++s) {
                h[s] = h[s] + sp * (a[s] * h[s] + cf * sB[tl][s]);
                y += sC[tl][s] * h[s];
            }
            float sig = 1.f / (1.f + expf(-zv));
            Y[idx] = y * (zv * sig);                                    // SiLU gate
        }
    }
}

extern "C" void kernel_launch(void* const* d_in, const int* in_sizes, int n_in,
                              void* d_out, int out_size, void* d_ws, size_t ws_size,
                              hipStream_t stream) {
    const float* x     = (const float*)d_in[0];
    const float* W_in  = (const float*)d_in[1];
    const float* b_in  = (const float*)d_in[2];
    const float* A     = (const float*)d_in[3];
    const float* W_B   = (const float*)d_in[4];
    const float* b_B   = (const float*)d_in[5];
    const float* W_C   = (const float*)d_in[6];
    const float* b_C   = (const float*)d_in[7];
    const float* W_dt  = (const float*)d_in[8];
    const float* b_dt  = (const float*)d_in[9];
    const float* W_out = (const float*)d_in[10];
    const float* b_out = (const float*)d_in[11];
    const float* gamma = (const float*)d_in[12];
    const float* beta  = (const float*)d_in[13];
    float* out = (float*)d_out;

    // workspace carve-out (256B aligned)
    char* base = (char*)d_ws;
    size_t off = 0;
    auto carve = [&](size_t bytes) -> char* {
        char* p = base + off;
        off += (bytes + 255) & ~(size_t)255;
        return p;
    };
    float* xn    = (float*)carve((size_t)NTOK * D_MODEL * 4);
    float* zbuf  = (float*)carve((size_t)NTOK * D_INNER * 4);
    float* dtp   = (float*)carve((size_t)NTOK * D_INNER * 4);
    float* Ybuf  = (float*)carve((size_t)NTOK * D_INNER * 4);
    bf16*  Ahi   = (bf16*) carve((size_t)NTOK * D_MODEL * 2);
    bf16*  Alo   = (bf16*) carve((size_t)NTOK * D_MODEL * 2);
    bf16*  Wzhi  = (bf16*) carve((size_t)D_MODEL * D_INNER * 2);
    bf16*  Wzlo  = (bf16*) carve((size_t)D_MODEL * D_INNER * 2);
    bf16*  Wdhi  = (bf16*) carve((size_t)D_MODEL * D_INNER * 2);
    bf16*  Wdlo  = (bf16*) carve((size_t)D_MODEL * D_INNER * 2);
    bf16*  Wohi  = (bf16*) carve((size_t)D_INNER * D_MODEL * 2);
    bf16*  Wolo  = (bf16*) carve((size_t)D_INNER * D_MODEL * 2);
    bf16*  Yhi   = (bf16*) carve((size_t)NTOK * D_INNER * 2);
    bf16*  Ylo   = (bf16*) carve((size_t)NTOK * D_INNER * 2);
    float* Bm    = (float*)carve((size_t)NTOK * D_STATE * 4);
    float* Cm    = (float*)carve((size_t)NTOK * D_STATE * 4);
    float* xsum  = (float*)carve((size_t)NTOK * 4);
    float* wsum  = (float*)carve((size_t)(D_MODEL + 1) * 4);

    // 1) LayerNorm
    k_ln<<<NTOK, 256, 0, stream>>>(x, gamma, beta, xn);

    // 2) pack activations + weights into WMMA fragment layouts (hi/lo bf16)
    {
        int tot = (NTOK / 16) * (D_MODEL / 32) * 32;
        k_pack_a<<<(tot + 255) / 256, 256, 0, stream>>>(xn, Ahi, Alo, NTOK, D_MODEL);
    }
    {
        int tot = (D_MODEL / 32) * (D_INNER / 16) * 32;
        k_pack_w<<<(tot + 255) / 256, 256, 0, stream>>>(W_in, Wzhi, Wzlo, D_MODEL, D_INNER, 2 * D_INNER, 0);
        k_pack_w<<<(tot + 255) / 256, 256, 0, stream>>>(W_dt, Wdhi, Wdlo, D_MODEL, D_INNER, D_INNER, 0);
    }
    {
        int tot = (D_INNER / 32) * (D_MODEL / 16) * 32;
        k_pack_w<<<(tot + 255) / 256, 256, 0, stream>>>(W_out, Wohi, Wolo, D_INNER, D_MODEL, D_MODEL, 0);
    }

    // 3) x_sum weight reduction + small projections
    k_reduce_win<<<(D_MODEL + 255) / 256, 256, 0, stream>>>(W_in, b_in, wsum);
    k_bcx<<<NTOK, 64, 0, stream>>>(xn, W_B, b_B, W_C, b_C, wsum, Bm, Cm, xsum);

    // 4) big WMMA GEMMs: z = xn@W_in[:, :2048] + b_in[:2048]; dtpre = xn@W_dt + b_dt
    {
        int waves = (NTOK / 32) * (D_INNER / 64);
        k_gemm<<<waves / 8, 256, 0, stream>>>(Ahi, Alo, Wzhi, Wzlo, zbuf,
                                              NTOK, D_INNER, D_MODEL, b_in, nullptr);
        k_gemm<<<waves / 8, 256, 0, stream>>>(Ahi, Alo, Wdhi, Wdlo, dtp,
                                              NTOK, D_INNER, D_MODEL, b_dt, nullptr);
    }

    // 5) selective scan (fused softplus + SiLU gate)
    k_scan<<<dim3(D_INNER / 256, BATCH), 256, 0, stream>>>(dtp, zbuf, Bm, Cm, xsum, A, Ybuf);

    // 6) pack Y, output GEMM with fused bias + residual
    {
        int tot = (NTOK / 16) * (D_INNER / 32) * 32;
        k_pack_a<<<(tot + 255) / 256, 256, 0, stream>>>(Ybuf, Yhi, Ylo, NTOK, D_INNER);
    }
    {
        int waves = (NTOK / 32) * (D_MODEL / 64);
        k_gemm<<<waves / 8, 256, 0, stream>>>(Yhi, Ylo, Wohi, Wolo, out,
                                              NTOK, D_MODEL, D_INNER, b_out, x);
    }
}
